// MLAAttention_50697793962492
// MI455X (gfx1250) — matrix operand order repaired and checked by reading, not compiled, not run on hardware
//
#include <hip/hip_runtime.h>

typedef __attribute__((ext_vector_type(16))) _Float16 v16h;
typedef __attribute__((ext_vector_type(8)))  float    v8f;

// ---------- helpers ----------

__device__ __forceinline__ void load8(v16h& d, int j0, const _Float16* p) {
  union { uint4 u; _Float16 h[8]; } t;
  t.u = *(const uint4*)p;
#pragma unroll
  for (int i = 0; i < 8; ++i) d[j0 + i] = t.h[i];
}

__device__ __forceinline__ v8f wmma_f16(v16h a, v16h b, v8f c) {
  // D = A(16x32 f16) * B(32x16 f16) + C(16x16 f32)
  return __builtin_amdgcn_wmma_f32_16x16x32_f16(false, a, false, b, (short)0, c,
                                                false, false);
}

// Async global->LDS copy, 16B per lane, tracked by ASYNCcnt (CDNA5 TDM-lite path).
// LDS destination byte offset = low 32 bits of the generic pointer to __shared__.
__device__ __forceinline__ void async_copy_b128(_Float16* lds_dst, const _Float16* gsrc) {
  unsigned loff = (unsigned)(size_t)lds_dst;
  asm volatile("global_load_async_to_lds_b128 %0, %1, off"
               :: "v"(loff), "v"(gsrc) : "memory");
}

// ---------- conversion kernels ----------

__global__ void cvt_f16_kernel(_Float16* __restrict__ dst,
                               const float* __restrict__ src, int n) {
  int i = blockIdx.x * blockDim.x + threadIdx.x;
  if (i < n) dst[i] = (_Float16)src[i];
}

// src is [K][N] row-major f32; dst is [N][K] row-major f16 (transposed)
__global__ void cvt_transpose_kernel(_Float16* __restrict__ dst,
                                     const float* __restrict__ src,
                                     int K, int N) {
  int i = blockIdx.x * blockDim.x + threadIdx.x;
  if (i < K * N) {
    int k = i / N, n = i % N;
    dst[(size_t)n * K + k] = (_Float16)src[i];
  }
}

// ---------- QKV GEMM:  qkv[8192,2304] = Xh[8192,768] @ Wqkv ----------
// 256 threads = 8 waves, each wave computes a 32x32 tile (2x2 WMMA tiles).
// Output scattered: Q,K -> [B,H,T,D] f16 ; V -> [B,H,D,T] f16 (transposed).

__global__ void qkv_gemm_kernel(const _Float16* __restrict__ Xh,
                                const _Float16* __restrict__ WT, // [2304][768]
                                _Float16* __restrict__ Qg,
                                _Float16* __restrict__ Kg,
                                _Float16* __restrict__ Vtg) {
  const int lane = threadIdx.x & 31;
  const int wave = threadIdx.x >> 5;
  const int half = lane >> 4;
  const int l15  = lane & 15;
  const int m0 = blockIdx.y * 64 + (wave >> 2) * 32;
  const int n0 = blockIdx.x * 128 + (wave & 3) * 32;

  v8f acc00 = {}, acc01 = {}, acc10 = {}, acc11 = {};
  for (int kk = 0; kk < 768; kk += 32) {
    v16h a0, a1, b0, b1;
    const _Float16* p;
    p = Xh + (size_t)(m0 + l15) * 768 + kk + half * 8;
    load8(a0, 0, p); load8(a0, 8, p + 16);
    p = Xh + (size_t)(m0 + 16 + l15) * 768 + kk + half * 8;
    load8(a1, 0, p); load8(a1, 8, p + 16);
    p = WT + (size_t)(n0 + l15) * 768 + kk + half * 8;
    load8(b0, 0, p); load8(b0, 8, p + 16);
    p = WT + (size_t)(n0 + 16 + l15) * 768 + kk + half * 8;
    load8(b1, 0, p); load8(b1, 8, p + 16);
    acc00 = wmma_f16(a0, b0, acc00);
    acc01 = wmma_f16(a0, b1, acc01);
    acc10 = wmma_f16(a1, b0, acc10);
    acc11 = wmma_f16(a1, b1, acc11);
  }

#pragma unroll
  for (int ms = 0; ms < 2; ++ms) {
#pragma unroll
    for (int ns = 0; ns < 2; ++ns) {
      const v8f& acc = ms ? (ns ? acc11 : acc10) : (ns ? acc01 : acc00);
#pragma unroll
      for (int r = 0; r < 8; ++r) {
        int m = m0 + ms * 16 + r + half * 8;     // global row (b*T + t)
        int n = n0 + ns * 16 + l15;              // global col in [0,2304)
        int sel = n / 768, c = n % 768;
        int hh = c >> 6, d = c & 63;
        int b = m >> 12, t = m & 4095;
        size_t bh = (size_t)b * 12 + hh;
        _Float16 val = (_Float16)acc[r];
        if (sel == 0)      Qg[(bh * 4096 + t) * 64 + d] = val;
        else if (sel == 1) Kg[(bh * 4096 + t) * 64 + d] = val;
        else               Vtg[(bh * 64 + d) * 4096 + t] = val;
      }
    }
  }
}

// ---------- Flash attention ----------
// grid = B*H*(T/64) = 1536 blocks, 128 threads (4 waves).
// Each wave owns 16 query rows. 32-key K / V^T tiles are double-buffered in
// LDS and filled with global_load_async_to_lds_b128 (ASYNCcnt pipeline).

#define KT_PITCH 72   // 144 B  (16B-aligned rows)
#define VT_PITCH 40   // 80 B
#define PT_PITCH 40   // 80 B

__global__ void attn_kernel(const _Float16* __restrict__ Qg,
                            const _Float16* __restrict__ Kg,
                            const _Float16* __restrict__ Vtg,
                            _Float16* __restrict__ Yg) {
  __shared__ alignas(16) _Float16 Kt[2][32 * KT_PITCH];    // [buf][key][d]
  __shared__ alignas(16) _Float16 Vt[2][64 * VT_PITCH];    // [buf][d][key]
  __shared__ alignas(16) _Float16 Pt[4 * 16 * PT_PITCH];   // per-wave [m][key]

  const int tid  = threadIdx.x;
  const int lane = tid & 31;
  const int wave = tid >> 5;
  const int half = lane >> 4;
  const int l15  = lane & 15;

  const int bh = blockIdx.x >> 6;      // 0..23
  const int qc = blockIdx.x & 63;
  const int b  = bh / 12;
  const int h  = bh % 12;
  const int q0 = qc * 64;

  const size_t qkbase = (size_t)bh * 4096 * 64;
  const size_t vbase  = (size_t)bh * 64 * 4096;

  // per-thread slices for cooperative tile staging (4 async b128 per thread)
  const int krow = tid >> 2, kcol = (tid & 3) * 16;   // K: 32 keys x 64 d
  const int vrow = tid >> 1, vcol = (tid & 1) * 16;   // V^T: 64 d x 32 keys

  auto issue_tile = [&](int k0, int buf) {
    const _Float16* gk = Kg + qkbase + (size_t)(k0 + krow) * 64 + kcol;
    _Float16* lk = &Kt[buf][krow * KT_PITCH + kcol];
    async_copy_b128(lk,     gk);
    async_copy_b128(lk + 8, gk + 8);
    const _Float16* gv = Vtg + vbase + (size_t)vrow * 4096 + k0 + vcol;
    _Float16* lv = &Vt[buf][vrow * VT_PITCH + vcol];
    async_copy_b128(lv,     gv);
    async_copy_b128(lv + 8, gv + 8);
  };

  issue_tile(0, 0);   // prefetch tile 0 while we fetch Q fragments

  // Q A-fragments for this wave's 16 rows (K-dim 64 -> two 32-chunks)
  const _Float16* qp = Qg + qkbase + (size_t)(q0 + wave * 16 + l15) * 64;
  v16h qa0, qa1;
  load8(qa0, 0, qp + half * 8);       load8(qa0, 8, qp + half * 8 + 16);
  load8(qa1, 0, qp + 32 + half * 8);  load8(qa1, 8, qp + 32 + half * 8 + 16);

  v8f oacc[4];
#pragma unroll
  for (int i = 0; i < 4; ++i) { v8f z = {}; oacc[i] = z; }
  float mrun[8], lrun[8];
#pragma unroll
  for (int r = 0; r < 8; ++r) { mrun[r] = -1.0e30f; lrun[r] = 0.0f; }

  _Float16* ptw = Pt + wave * 16 * PT_PITCH;
  const float scale = 0.125f;   // 1/sqrt(64)

  for (int it = 0; it < 128; ++it) {
    const int buf = it & 1;
    if (it + 1 < 128) {
      issue_tile((it + 1) * 32, buf ^ 1);
      // 4 newer ops outstanding; older 4 (this tile) complete in order
      asm volatile("s_wait_asynccnt 0x4" ::: "memory");
    } else {
      asm volatile("s_wait_asynccnt 0x0" ::: "memory");
    }
    __syncthreads();   // every wave's portion of tile `it` is in LDS

    const _Float16* KtB = Kt[buf];
    const _Float16* VtB = Vt[buf];

    // S = Q K^T  (16 rows x 32 keys)
    v8f s0 = {}, s1 = {};
    {
      v16h kb;
      const _Float16* kp;
      kp = KtB + l15 * KT_PITCH + half * 8;              // keys 0-15, d 0-31
      load8(kb, 0, kp); load8(kb, 8, kp + 16);
      s0 = wmma_f16(qa0, kb, s0);
      kp = KtB + l15 * KT_PITCH + 32 + half * 8;         // keys 0-15, d 32-63
      load8(kb, 0, kp); load8(kb, 8, kp + 16);
      s0 = wmma_f16(qa1, kb, s0);
      kp = KtB + (16 + l15) * KT_PITCH + half * 8;       // keys 16-31, d 0-31
      load8(kb, 0, kp); load8(kb, 8, kp + 16);
      s1 = wmma_f16(qa0, kb, s1);
      kp = KtB + (16 + l15) * KT_PITCH + 32 + half * 8;  // keys 16-31, d 32-63
      load8(kb, 0, kp); load8(kb, 8, kp + 16);
      s1 = wmma_f16(qa1, kb, s1);
    }

    // Online softmax: row = r + 8*half, columns live in lanes l15 of this half.
#pragma unroll
    for (int r = 0; r < 8; ++r) {
      float a = s0[r] * scale;
      float c = s1[r] * scale;
      float mx = fmaxf(a, c);
#pragma unroll
      for (int m = 8; m >= 1; m >>= 1) mx = fmaxf(mx, __shfl_xor(mx, m, 32));
      float mnew  = fmaxf(mrun[r], mx);
      float alpha = __expf(mrun[r] - mnew);
      float p0 = __expf(a - mnew);
      float p1 = __expf(c - mnew);
      float rs = p0 + p1;
#pragma unroll
      for (int m = 8; m >= 1; m >>= 1) rs += __shfl_xor(rs, m, 32);
      lrun[r] = lrun[r] * alpha + rs;
      mrun[r] = mnew;
#pragma unroll
      for (int ns = 0; ns < 4; ++ns) oacc[ns][r] *= alpha;
      ptw[(r + half * 8) * PT_PITCH + l15]      = (_Float16)p0;
      ptw[(r + half * 8) * PT_PITCH + l15 + 16] = (_Float16)p1;
    }
    __syncthreads();   // publish P across the wave halves

    // O += P (16x32) @ V (32x64)
    v16h pa;
    const _Float16* pp = ptw + l15 * PT_PITCH + half * 8;
    load8(pa, 0, pp); load8(pa, 8, pp + 16);
#pragma unroll
    for (int ns = 0; ns < 4; ++ns) {
      v16h vb;
      const _Float16* vp = VtB + (ns * 16 + l15) * VT_PITCH + half * 8;
      load8(vb, 0, vp); load8(vb, 8, vp + 16);
      oacc[ns] = wmma_f16(pa, vb, oacc[ns]);
    }
    __syncthreads();   // all reads of buf done before tile it+2 overwrites it
  }

  // Normalize and write Y in [B,T,C] f16
  float inv[8];
#pragma unroll
  for (int r = 0; r < 8; ++r) inv[r] = 1.0f / lrun[r];
#pragma unroll
  for (int ns = 0; ns < 4; ++ns) {
#pragma unroll
    for (int r = 0; r < 8; ++r) {
      int t   = q0 + wave * 16 + r + half * 8;
      int col = h * 64 + ns * 16 + l15;
      Yg[((size_t)b * 4096 + t) * 768 + col] = (_Float16)(oacc[ns][r] * inv[r]);
    }
  }
}

// ---------- projection GEMM: out[8192,768] (f32) = Y @ Wproj ----------

__global__ void proj_gemm_kernel(const _Float16* __restrict__ Yh,
                                 const _Float16* __restrict__ WT, // [768][768]
                                 float* __restrict__ out) {
  const int lane = threadIdx.x & 31;
  const int wave = threadIdx.x >> 5;
  const int half = lane >> 4;
  const int l15  = lane & 15;
  const int m0 = blockIdx.y * 64 + (wave >> 2) * 32;
  const int n0 = blockIdx.x * 128 + (wave & 3) * 32;

  v8f acc00 = {}, acc01 = {}, acc10 = {}, acc11 = {};
  for (int kk = 0; kk < 768; kk += 32) {
    v16h a0, a1, b0, b1;
    const _Float16* p;
    p = Yh + (size_t)(m0 + l15) * 768 + kk + half * 8;
    load8(a0, 0, p); load8(a0, 8, p + 16);
    p = Yh + (size_t)(m0 + 16 + l15) * 768 + kk + half * 8;
    load8(a1, 0, p); load8(a1, 8, p + 16);
    p = WT + (size_t)(n0 + l15) * 768 + kk + half * 8;
    load8(b0, 0, p); load8(b0, 8, p + 16);
    p = WT + (size_t)(n0 + 16 + l15) * 768 + kk + half * 8;
    load8(b1, 0, p); load8(b1, 8, p + 16);
    acc00 = wmma_f16(a0, b0, acc00);
    acc01 = wmma_f16(a0, b1, acc01);
    acc10 = wmma_f16(a1, b0, acc10);
    acc11 = wmma_f16(a1, b1, acc11);
  }

#pragma unroll
  for (int ms = 0; ms < 2; ++ms) {
#pragma unroll
    for (int ns = 0; ns < 2; ++ns) {
      const v8f& acc = ms ? (ns ? acc11 : acc10) : (ns ? acc01 : acc00);
#pragma unroll
      for (int r = 0; r < 8; ++r) {
        int m = m0 + ms * 16 + r + half * 8;
        int n = n0 + ns * 16 + l15;
        out[(size_t)m * 768 + n] = acc[r];
      }
    }
  }
}

// ---------- host launcher ----------

extern "C" void kernel_launch(void* const* d_in, const int* in_sizes, int n_in,
                              void* d_out, int out_size, void* d_ws, size_t ws_size,
                              hipStream_t stream) {
  const float* x     = (const float*)d_in[0];   // [2,4096,768]
  const float* wqkv  = (const float*)d_in[1];   // [768,2304]
  const float* wproj = (const float*)d_in[2];   // [768,768]
  float* out = (float*)d_out;

  const size_t M = 8192, C = 768, HD = (size_t)24 * 4096 * 64; // B*H*T*D

  _Float16* Xh     = (_Float16*)d_ws;           // M*C
  _Float16* WqkvT  = Xh + M * C;                // [2304][768]
  _Float16* WprojT = WqkvT + (size_t)2304 * 768;
  _Float16* Qg     = WprojT + (size_t)768 * 768; // [B,H,T,D]
  _Float16* Kg     = Qg + HD;                    // [B,H,T,D]
  _Float16* Vtg    = Kg + HD;                    // [B,H,D,T]
  _Float16* Yg     = Vtg + HD;                   // [B,T,C]

  (void)in_sizes; (void)n_in; (void)out_size; (void)ws_size;

  cvt_f16_kernel<<<dim3((unsigned)((M * C + 255) / 256)), 256, 0, stream>>>(
      Xh, x, (int)(M * C));
  cvt_transpose_kernel<<<dim3((768 * 2304 + 255) / 256), 256, 0, stream>>>(
      WqkvT, wqkv, 768, 2304);
  cvt_transpose_kernel<<<dim3((768 * 768 + 255) / 256), 256, 0, stream>>>(
      WprojT, wproj, 768, 768);

  qkv_gemm_kernel<<<dim3(18, 128), 256, 0, stream>>>(Xh, WqkvT, Qg, Kg, Vtg);
  attn_kernel<<<dim3(1536), 128, 0, stream>>>(Qg, Kg, Vtg, Yg);
  proj_gemm_kernel<<<dim3(6, 128), 256, 0, stream>>>(Yg, WprojT, out);
}
